// FeaturePnP_23759759082074
// MI455X (gfx1250) — matrix operand order repaired and checked by reading, not compiled, not run on hardware
//
#include <hip/hip_runtime.h>
#include <hip/hip_bf16.h>
#include <math.h>

typedef __attribute__((ext_vector_type(16))) _Float16 v16h;
typedef __attribute__((ext_vector_type(8)))  float    v8f;

#define WAVES 8
#define ITERS 10

__device__ __forceinline__ int iclampi(int v, int lo, int hi) {
  return v < lo ? lo : (v > hi ? hi : v);
}

// ---------------------------------------------------------------------------
// State layout in ws (floats):
//  [0..8]  R (row-major)   [9..11] t      [12] lam   [13] prev
//  [14..22] R_new          [23..25] t_new [26] delta-NaN flag
// ---------------------------------------------------------------------------

__global__ void k_init(const float* __restrict__ q, const float* __restrict__ r,
                       float* __restrict__ ST) {
  if (threadIdx.x == 0) {
    // inv(r) via Gauss-Jordan with partial pivoting
    float a[4][8];
    for (int i = 0; i < 4; ++i)
      for (int j = 0; j < 4; ++j) {
        a[i][j]     = r[i * 4 + j];
        a[i][j + 4] = (i == j) ? 1.f : 0.f;
      }
    for (int k = 0; k < 4; ++k) {
      int piv = k; float mx = fabsf(a[k][k]);
      for (int i = k + 1; i < 4; ++i) { float v = fabsf(a[i][k]); if (v > mx) { mx = v; piv = i; } }
      if (piv != k)
        for (int j = 0; j < 8; ++j) { float tmp = a[k][j]; a[k][j] = a[piv][j]; a[piv][j] = tmp; }
      float d = a[k][k];
      for (int j = 0; j < 8; ++j) a[k][j] /= d;
      for (int i = 0; i < 4; ++i)
        if (i != k) { float f = a[i][k]; for (int j = 0; j < 8; ++j) a[i][j] -= f * a[k][j]; }
    }
    // rel = q @ inv(r)
    float rel[16];
    for (int i = 0; i < 4; ++i)
      for (int j = 0; j < 4; ++j) {
        float s = 0.f;
        for (int k = 0; k < 4; ++k) s += q[i * 4 + k] * a[k][j + 4];
        rel[i * 4 + j] = s;
      }
    for (int i = 0; i < 3; ++i)
      for (int j = 0; j < 3; ++j) ST[i * 3 + j] = rel[i * 4 + j];
    ST[9]  = rel[12]; ST[10] = rel[13]; ST[11] = rel[14];   // t = rel[3,:3]
    ST[12] = 0.01f;                                          // LAMBDA0
    ST[13] = __builtin_inff();                               // prev
    ST[26] = 0.f;                                            // nan flag
  }
}

// pts0 = from_homogeneous(pts_h @ r^T)
__global__ void k_pts0(const float* __restrict__ p3d, const float* __restrict__ r,
                       float* __restrict__ pts0, int N) {
  int n = blockIdx.x * blockDim.x + threadIdx.x;
  if (n < N) {
    float X = p3d[3 * n + 0], Y = p3d[3 * n + 1], Z = p3d[3 * n + 2];
    float c0 = r[0] * X + r[1] * Y + r[2]  * Z + r[3];
    float c1 = r[4] * X + r[5] * Y + r[6]  * Z + r[7];
    float c2 = r[8] * X + r[9] * Y + r[10] * Z + r[11];
    float w  = r[12] * X + r[13] * Y + r[14] * Z + r[15];
    pts0[3 * n + 0] = c0 / w;
    pts0[3 * n + 1] = c1 / w;
    pts0[3 * n + 2] = c2 / w;
  }
}

// ---------------------------------------------------------------------------
// Main per-iteration reduce: per 16-point tile compute 6 channel reductions
// via v_wmma_f32_16x16x32_f16 Gram products (diagonals), then partial H/g/cost.
// Requires N % 16 == 0, C % (32*WAVES) == 0 (harness: N=8192, C=1024).
// ---------------------------------------------------------------------------
__global__ __launch_bounds__(256) void k_reduce(
    const float* __restrict__ qf, const float* __restrict__ rf,
    const float* __restrict__ K, const int* __restrict__ ratio_p,
    const float* __restrict__ p2d, const float* __restrict__ pts0,
    const float* __restrict__ ST, float* __restrict__ part,
    int C, int H, int W) {
  __shared__ _Float16 stg[WAVES][3][16][32];   // [wave][e/gx/gy][point][chan]
  __shared__ float    accL[6][256];
  __shared__ float    Mrow[16][12];            // M0[6], M1[6] per point
  __shared__ int      pxl[16][4];              // ix, iy, ix0, iy0
  __shared__ float    contrib[16][44];

  const int tid  = threadIdx.x;
  const int lane = tid & 31;
  const int wv   = tid >> 5;
  const int HWp  = H * W;
  const float ratio = (float)(*ratio_p);
  const int base = blockIdx.x * 16;

  if (tid < 16) {
    int n = base + tid;
    float p0x = pts0[3 * n], p0y = pts0[3 * n + 1], p0z = pts0[3 * n + 2];
    // pts1 = pts0 @ R + t  (row-vector convention)
    float x = p0x * ST[0] + p0y * ST[3] + p0z * ST[6] + ST[9];
    float y = p0x * ST[1] + p0y * ST[4] + p0z * ST[7] + ST[10];
    float z = p0x * ST[2] + p0y * ST[5] + p0z * ST[8] + ST[11];
    float hx = K[0] * x + K[1] * y + K[2] * z;
    float hy = K[3] * x + K[4] * y + K[5] * z;
    float hz = K[6] * x + K[7] * y + K[8] * z;
    int ix = iclampi((int)floorf(hx / hz / ratio), 0, W - 1);
    int iy = iclampi((int)floorf(hy / hz / ratio), 0, H - 1);
    int ix0 = iclampi((int)floorf(p2d[2 * n]     / ratio), 0, W - 1);
    int iy0 = iclampi((int)floorf(p2d[2 * n + 1] / ratio), 0, H - 1);
    pxl[tid][0] = ix; pxl[tid][1] = iy; pxl[tid][2] = ix0; pxl[tid][3] = iy0;
    // M = J_h_p @ J_p_T  (2x6)
    float fx = K[0], fy = K[4];
    float iz = 1.f / z;
    float a00 = fx * iz / ratio, a02 = -fx * x * iz * iz / ratio;
    float a11 = fy * iz / ratio, a12 = -fy * y * iz * iz / ratio;
    // -skew(p1) rows: (0,z,-y) (-z,0,x) (y,-x,0)
    float M0[6], M1[6];
    M0[0] = a00; M0[1] = 0.f; M0[2] = a02;
    M0[3] = a02 * y;
    M0[4] = a00 * z - a02 * x;
    M0[5] = -a00 * y;
    M1[0] = 0.f; M1[1] = a11; M1[2] = a12;
    M1[3] = -a11 * z + a12 * y;
    M1[4] = -a12 * x;
    M1[5] = a11 * x;
    for (int k = 0; k < 6; ++k) { Mrow[tid][k] = M0[k]; Mrow[tid][6 + k] = M1[k]; }
  }
  __syncthreads();

  v8f axx = {}, axy = {}, ayy = {}, axe = {}, aye = {}, aee = {};
  const int m     = lane & 15;
  const int koffA = (lane >> 4) * 8;
  const int koffB = (lane >> 4) * 16;
  const int perWave = C / (32 * WAVES);

  for (int it = 0; it < perWave; ++it) {
    // ---- stage: lane handles channel (cbase+lane) for all 16 points ----
    int c = (it * WAVES + wv) * 32 + lane;
    const float* qc = qf + (size_t)c * HWp;
    const float* rc = rf + (size_t)c * HWp;
    for (int n = 0; n < 16; ++n) {
      int ix = pxl[n][0], iy = pxl[n][1];
      float sv[3][3];
#pragma unroll
      for (int dy = 0; dy < 3; ++dy)
#pragma unroll
        for (int dx = 0; dx < 3; ++dx) {
          int yy = iy + dy - 1, xx = ix + dx - 1;
          sv[dy][dx] = (yy >= 0 && yy < H && xx >= 0 && xx < W) ? qc[yy * W + xx] : 0.f;
        }
      float gxv = (sv[0][2] - sv[0][0]) + 2.f * (sv[1][2] - sv[1][0]) + (sv[2][2] - sv[2][0]);
      float gyv = (sv[2][0] - sv[0][0]) + 2.f * (sv[2][1] - sv[0][1]) + (sv[2][2] - sv[0][2]);
      float ev  = sv[1][1] - rc[pxl[n][3] * W + pxl[n][2]];
      stg[wv][0][n][lane] = (_Float16)ev;
      stg[wv][1][n][lane] = (_Float16)gxv;
      stg[wv][2][n][lane] = (_Float16)gyv;
    }
    __syncthreads();

    // ---- build WMMA operands per gfx1250 16-bit A(16x32)/B(32x16) layouts ----
    v16h Ae, Agx, Agy, Be, Bgx, Bgy;
#pragma unroll
    for (int j = 0; j < 16; ++j) {
      int ca = koffA + j + ((j >= 8) ? 8 : 0);  // A: K in {koffA..+7, koffA+16..+23}
      Ae[j]  = stg[wv][0][m][ca];
      Agx[j] = stg[wv][1][m][ca];
      Agy[j] = stg[wv][2][m][ca];
      int cb = koffA + j + ((j >= 8) ? 8 : 0);  // B: K in {koffA..+7, koffA+16..+23}
      Be[j]  = stg[wv][0][m][cb];
      Bgx[j] = stg[wv][1][m][cb];
      Bgy[j] = stg[wv][2][m][cb];
    }
    axx = __builtin_amdgcn_wmma_f32_16x16x32_f16(false, Agx, false, Bgx, (short)0, axx, false, false);
    axy = __builtin_amdgcn_wmma_f32_16x16x32_f16(false, Agx, false, Bgy, (short)0, axy, false, false);
    ayy = __builtin_amdgcn_wmma_f32_16x16x32_f16(false, Agy, false, Bgy, (short)0, ayy, false, false);
    axe = __builtin_amdgcn_wmma_f32_16x16x32_f16(false, Agx, false, Be,  (short)0, axe, false, false);
    aye = __builtin_amdgcn_wmma_f32_16x16x32_f16(false, Agy, false, Be,  (short)0, aye, false, false);
    aee = __builtin_amdgcn_wmma_f32_16x16x32_f16(false, Ae,  false, Be,  (short)0, aee, false, false);
    __syncthreads();
  }

  // ---- deterministic cross-wave accumulation (fixed order) ----
  for (int w = 0; w < WAVES; ++w) {
    if (wv == w) {
#pragma unroll
      for (int r = 0; r < 8; ++r) {
        int idx = r * 32 + lane;
        if (w == 0) {
          accL[0][idx] = axx[r]; accL[1][idx] = axy[r]; accL[2][idx] = ayy[r];
          accL[3][idx] = axe[r]; accL[4][idx] = aye[r]; accL[5][idx] = aee[r];
        } else {
          accL[0][idx] += axx[r]; accL[1][idx] += axy[r]; accL[2][idx] += ayy[r];
          accL[3][idx] += axe[r]; accL[4][idx] += aye[r]; accL[5][idx] += aee[r];
        }
      }
    }
    __syncthreads();
  }

  // ---- diagonal extraction + per-point H/g/cost contributions ----
  if (tid < 16) {
    int r  = (tid < 8) ? tid : tid - 8;
    int ln = (tid < 8) ? tid : tid + 16;   // D layout: lanes16-31 hold M=8..15, N=lane-16
    int idx = r * 32 + ln;
    float Sxx = accL[0][idx], Sxy = accL[1][idx], Syy = accL[2][idx];
    float Sxe = accL[3][idx], Sye = accL[4][idx], See = accL[5][idx];
    float M0[6], M1[6];
    for (int k = 0; k < 6; ++k) { M0[k] = Mrow[tid][k]; M1[k] = Mrow[tid][6 + k]; }
    for (int k = 0; k < 6; ++k)
      for (int l = 0; l < 6; ++l)
        contrib[tid][k * 6 + l] = Sxx * M0[k] * M0[l]
                                + Sxy * (M0[k] * M1[l] + M1[k] * M0[l])
                                + Syy * M1[k] * M1[l];
    for (int k = 0; k < 6; ++k) contrib[tid][36 + k] = Sxe * M0[k] + Sye * M1[k];
    contrib[tid][42] = See;
  }
  __syncthreads();
  if (tid < 43) {
    float s = 0.f;
    for (int n = 0; n < 16; ++n) s += contrib[n][tid];
    part[(size_t)blockIdx.x * 44 + tid] = s;
  }
}

// ---------------------------------------------------------------------------
// Reduce partials, build LM system, solve 6x6, so3exp update -> trial pose.
// ---------------------------------------------------------------------------
__global__ void k_solve(float* __restrict__ ST, const float* __restrict__ part,
                        int nb, int N, int iter) {
  __shared__ float red[43];
  int tid = threadIdx.x;
  if (tid < 43) {
    float s = 0.f;
    for (int b = 0; b < nb; ++b) s += part[(size_t)b * 44 + tid];
    red[tid] = s;
  }
  __syncthreads();
  if (tid == 0) {
    float costm = red[42] / (float)N;
    float prev  = ST[13];
    if (iter == 0) prev = costm;
    ST[13] = prev;
    float lam = ST[12];
    float A[6][7];
    for (int r = 0; r < 6; ++r) {
      for (int c = 0; c < 6; ++c) A[r][c] = red[r * 6 + c];
      A[r][r] += (red[r * 6 + r] + 1e-9f) * lam;  // Hs + D*lam
      A[r][6] = -red[36 + r];                      // solve A x = -g
    }
    for (int k = 0; k < 6; ++k) {
      int piv = k; float mx = fabsf(A[k][k]);
      for (int r = k + 1; r < 6; ++r) { float v = fabsf(A[r][k]); if (v > mx) { mx = v; piv = r; } }
      if (piv != k)
        for (int c = 0; c < 7; ++c) { float t2 = A[k][c]; A[k][c] = A[piv][c]; A[piv][c] = t2; }
      float d = A[k][k];
      for (int c = k; c < 7; ++c) A[k][c] /= d;
      for (int r = 0; r < 6; ++r)
        if (r != k) { float f = A[r][k]; for (int c = k; c < 7; ++c) A[r][c] -= f * A[k][c]; }
    }
    float delta[6]; bool nanf = false;
    for (int r = 0; r < 6; ++r) { delta[r] = A[r][6]; nanf = nanf || (delta[r] != delta[r]); }
    ST[26] = nanf ? 1.f : 0.f;
    float wx = delta[3], wy = delta[4], wz = delta[5];
    float th2 = wx * wx + wy * wy + wz * wz + 1e-12f;
    float th  = sqrtf(th2);
    float Ac  = sinf(th) / th;
    float Bc  = (1.f - cosf(th)) / th2;
    float Wm[9] = { 0.f, -wz, wy,  wz, 0.f, -wx,  -wy, wx, 0.f };
    float W2[9];
    for (int i = 0; i < 3; ++i)
      for (int j = 0; j < 3; ++j) {
        float s = 0.f;
        for (int k = 0; k < 3; ++k) s += Wm[i * 3 + k] * Wm[k * 3 + j];
        W2[i * 3 + j] = s;
      }
    float dr[9];
    for (int i = 0; i < 9; ++i) dr[i] = ((i % 4 == 0) ? 1.f : 0.f) + Ac * Wm[i] + Bc * W2[i];
    // R_new = dr @ R ; t_new = dr @ t + dt
    for (int i = 0; i < 3; ++i)
      for (int j = 0; j < 3; ++j) {
        float s = 0.f;
        for (int k = 0; k < 3; ++k) s += dr[i * 3 + k] * ST[k * 3 + j];
        ST[14 + i * 3 + j] = s;
      }
    for (int i = 0; i < 3; ++i)
      ST[23 + i] = dr[i * 3] * ST[9] + dr[i * 3 + 1] * ST[10] + dr[i * 3 + 2] * ST[11] + delta[i];
  }
}

// ---------------------------------------------------------------------------
// Trial cost: npts = pts0 @ R_new^T + t_new. One wave per point.
// ---------------------------------------------------------------------------
__global__ __launch_bounds__(256) void k_trial(
    const float* __restrict__ qf, const float* __restrict__ rf,
    const float* __restrict__ K, const int* __restrict__ ratio_p,
    const float* __restrict__ p2d, const float* __restrict__ pts0,
    const float* __restrict__ ST, float* __restrict__ part2,
    int C, int H, int W) {
  __shared__ float wc[8];
  int tid = threadIdx.x, lane = tid & 31, wv = tid >> 5;
  int p = blockIdx.x * 8 + wv;
  const float ratio = (float)(*ratio_p);
  const int HWp = H * W;
  float p0x = pts0[3 * p], p0y = pts0[3 * p + 1], p0z = pts0[3 * p + 2];
  // note reference uses R_new.T here: comp j = dot(Rnew row j, p0) + tnew[j]
  float x = ST[14] * p0x + ST[15] * p0y + ST[16] * p0z + ST[23];
  float y = ST[17] * p0x + ST[18] * p0y + ST[19] * p0z + ST[24];
  float z = ST[20] * p0x + ST[21] * p0y + ST[22] * p0z + ST[25];
  float hx = K[0] * x + K[1] * y + K[2] * z;
  float hy = K[3] * x + K[4] * y + K[5] * z;
  float hz = K[6] * x + K[7] * y + K[8] * z;
  int jx = iclampi((int)floorf(hx / hz / ratio), 0, W - 1);
  int jy = iclampi((int)floorf(hy / hz / ratio), 0, H - 1);
  int ix0 = iclampi((int)floorf(p2d[2 * p]     / ratio), 0, W - 1);
  int iy0 = iclampi((int)floorf(p2d[2 * p + 1] / ratio), 0, H - 1);
  int offq = jy * W + jx, offr = iy0 * W + ix0;
  float acc = 0.f;
  for (int c = lane; c < C; c += 32) {
    float e = qf[(size_t)c * HWp + offq] - rf[(size_t)c * HWp + offr];
    acc += e * e;
  }
  for (int off = 16; off; off >>= 1) acc += __shfl_down(acc, off, 32);
  if (lane == 0) wc[wv] = acc;
  __syncthreads();
  if (tid == 0) {
    float s = 0.f;
    for (int i = 0; i < 8; ++i) s += wc[i];
    part2[blockIdx.x] = s;
  }
}

__global__ void k_accept(float* __restrict__ ST, const float* __restrict__ part2,
                         int nb2, int N) {
  __shared__ float red[64];
  int tid = threadIdx.x;
  if (tid < 64) {
    int per = nb2 / 64;
    float s = 0.f;
    for (int i = 0; i < per; ++i) s += part2[tid * per + i];
    red[tid] = s;
  }
  __syncthreads();
  if (tid == 0) {
    float s = 0.f;
    for (int i = 0; i < 64; ++i) s += red[i];
    float ncost = s / (float)N;
    bool bad   = (ncost != ncost) || (ST[26] != 0.f);
    float prev = ST[13];
    bool worse = ncost > prev;
    float lam  = ST[12];
    float lu   = lam * (worse ? 10.f : 0.1f);
    lu = fminf(fmaxf(lu, 1e-6f), 100.f);
    ST[12] = bad ? lam : lu;
    if (!(worse || bad)) {
      for (int i = 0; i < 9; ++i) ST[i] = ST[14 + i];
      for (int i = 0; i < 3; ++i) ST[9 + i] = ST[23 + i];
      ST[13] = ncost;
    }
  }
}

__global__ void k_out(const float* __restrict__ ST, float* __restrict__ out) {
  int t = threadIdx.x;
  if (t < 12) out[t] = ST[t];  // R (9, row-major) then t (3), contiguous in ST
}

extern "C" void kernel_launch(void* const* d_in, const int* in_sizes, int n_in,
                              void* d_out, int out_size, void* d_ws, size_t ws_size,
                              hipStream_t stream) {
  const float* q   = (const float*)d_in[0];
  const float* r   = (const float*)d_in[1];
  const float* p2d = (const float*)d_in[2];
  const float* p3d = (const float*)d_in[3];
  const float* qf  = (const float*)d_in[4];
  const float* rf  = (const float*)d_in[5];
  const float* K   = (const float*)d_in[6];
  const int* ratio = (const int*)d_in[7];
  const int N = in_sizes[2] / 2;
  const int H = 128, W = 128;
  const int C = in_sizes[4] / (H * W);

  float* wsf   = (float*)d_ws;
  float* ST    = wsf;                              // 64 floats state
  float* pts0  = wsf + 64;                         // 3*N
  const int NB1 = N / 16;
  float* part1 = pts0 + (size_t)3 * N;             // NB1*44
  const int NB2 = N / 8;
  float* part2 = part1 + (size_t)NB1 * 44;         // NB2

  k_init<<<1, 32, 0, stream>>>(q, r, ST);
  k_pts0<<<(N + 255) / 256, 256, 0, stream>>>(p3d, r, pts0, N);
  for (int i = 0; i < ITERS; ++i) {
    k_reduce<<<NB1, 256, 0, stream>>>(qf, rf, K, ratio, p2d, pts0, ST, part1, C, H, W);
    k_solve<<<1, 64, 0, stream>>>(ST, part1, NB1, N, i);
    k_trial<<<NB2, 256, 0, stream>>>(qf, rf, K, ratio, p2d, pts0, ST, part2, C, H, W);
    k_accept<<<1, 64, 0, stream>>>(ST, part2, NB2, N);
  }
  k_out<<<1, 16, 0, stream>>>(ST, (float*)d_out);
}
